// RWKV6Attention_CUDA_90314572300647
// MI455X (gfx1250) — compile-verified, matrix operand order checked by reading, not run on hardware
//
#include <hip/hip_runtime.h>
#include <hip/hip_bf16.h>

// ---------------- problem constants ----------------
constexpr int Bq   = 8;
constexpr int Tq   = 2048;
constexpr int Cq   = 1024;
constexpr int Hq   = 16;
constexpr int Sq   = 64;
constexpr int Fq   = 4 * Cq;          // 4096
constexpr int NTq  = Bq * Tq;         // 16384 rows

typedef __attribute__((ext_vector_type(16))) __bf16 v16bf;
typedef __attribute__((ext_vector_type(8)))  __bf16 v8bf;
typedef __attribute__((ext_vector_type(8)))  float  v8f;
typedef __attribute__((ext_vector_type(4)))  int    v4i;

#ifndef __has_builtin
#define __has_builtin(x) 0
#endif

// gfx1250 async direct-to-LDS path (ASYNCcnt-tracked), with safe fallback
#if defined(__gfx1250__) && __has_builtin(__builtin_amdgcn_global_load_async_to_lds_b128) && \
    __has_builtin(__builtin_amdgcn_s_wait_asynccnt)
#define USE_ASYNC_LDS_B128 1
#else
#define USE_ASYNC_LDS_B128 0
#endif

#if defined(__gfx1250__) && __has_builtin(__builtin_amdgcn_global_load_async_to_lds_b32) && \
    __has_builtin(__builtin_amdgcn_s_wait_asynccnt)
#define USE_ASYNC_LDS_B32 1
#else
#define USE_ASYNC_LDS_B32 0
#endif

// builtin signatures (from hipcc diagnostics):
//   b128: (v4i addrspace(1)*, v4i addrspace(3)*, imm int, imm int)
//   b32 : (int addrspace(1)*, int addrspace(3)*, imm int, imm int)
#define ASG_V4I(p) ((__attribute__((address_space(1))) v4i*)(void*)(p))
#define ASL_V4I(p) ((__attribute__((address_space(3))) v4i*)(void*)(p))
#define ASG_I32(p) ((__attribute__((address_space(1))) int*)(void*)(p))
#define ASL_I32(p) ((__attribute__((address_space(3))) int*)(void*)(p))

__device__ __forceinline__ __bf16 f2bf(float f) {
    union { float f; unsigned u; } c; c.f = f;
    unsigned r = c.u + 0x7fffu + ((c.u >> 16) & 1u);   // round-to-nearest-even
    unsigned short h = (unsigned short)(r >> 16);
    __bf16 b;
    __builtin_memcpy(&b, &h, 2);
    return b;
}

// ---------------- weight fp32 [K][N] -> bf16 [N][K] (tiled transpose) ----------------
__global__ void transpose_bf16_kernel(const float* __restrict__ in, __bf16* __restrict__ out,
                                      int K, int N) {
    __shared__ float tile[32][33];
    int n0 = blockIdx.x * 32, k0 = blockIdx.y * 32;
    for (int i = threadIdx.y; i < 32; i += 8)
        tile[i][threadIdx.x] = in[(size_t)(k0 + i) * N + n0 + threadIdx.x];
    __syncthreads();
    for (int i = threadIdx.y; i < 32; i += 8)
        out[(size_t)(n0 + i) * K + k0 + threadIdx.x] = f2bf(tile[threadIdx.x][i]);
}

// ---------------- LayerNorm: fp32 row -> bf16 row ----------------
__global__ void ln_kernel(const float* __restrict__ x, const float* __restrict__ g,
                          const float* __restrict__ bt, __bf16* __restrict__ out) {
    __shared__ float rs[256], rs2[256];
    int row = blockIdx.x, tid = threadIdx.x;
    const float* xr = x + (size_t)row * Cq;
    float v[4]; float s = 0.f, s2 = 0.f;
#pragma unroll
    for (int i = 0; i < 4; ++i) { float f = xr[tid + i * 256]; v[i] = f; s += f; s2 += f * f; }
    rs[tid] = s; rs2[tid] = s2; __syncthreads();
    for (int w = 128; w > 0; w >>= 1) {
        if (tid < w) { rs[tid] += rs[tid + w]; rs2[tid] += rs2[tid + w]; }
        __syncthreads();
    }
    float mean = rs[0] * (1.f / Cq);
    float var  = rs2[0] * (1.f / Cq) - mean * mean;
    float rstd = rsqrtf(var + 1e-5f);
#pragma unroll
    for (int i = 0; i < 4; ++i) {
        int c = tid + i * 256;
        out[(size_t)row * Cq + c] = f2bf((v[i] - mean) * rstd * g[c] + bt[c]);
    }
}

// ---------------- generic bf16 WMMA GEMM ----------------
// A: [M][K] bf16 row-major.  Bt: [N][K] bf16 (B transposed).
// mode 0: Out=fp32 acc; mode 1: Out=fp32 acc+Res; mode 2: Out=bf16 relu(acc)^2
#define TM 128
#define TN 128
#define TKs 32
#define LSTR 40   // padded LDS K-stride in halves (80B rows, conflict-free b128 frag loads)

__global__ __launch_bounds__(256) void gemm_bf16_kernel(
    const __bf16* __restrict__ A, const __bf16* __restrict__ Bt,
    void* __restrict__ Out, const float* __restrict__ Res,
    int M, int N, int K, int mode)
{
    __shared__ __bf16 sA[TM * LSTR];
    __shared__ __bf16 sB[TN * LSTR];

    const int tid  = threadIdx.x;
    const int wave = tid >> 5;
    const int lane = tid & 31;
    const int m0 = blockIdx.y * TM;
    const int n0 = blockIdx.x * TN;
    const int wm = (wave >> 1) * 32;   // 4 wave-rows of 32
    const int wn = (wave & 1) * 64;    // 2 wave-cols of 64

    v8f acc[2][4];
#pragma unroll
    for (int mt = 0; mt < 2; ++mt)
#pragma unroll
        for (int nt = 0; nt < 4; ++nt)
            acc[mt][nt] = v8f{0.f, 0.f, 0.f, 0.f, 0.f, 0.f, 0.f, 0.f};

    const int fr = lane & 15;            // fragment row/col within 16
    const int kb = (lane >> 4) * 8;      // K sub-offset 0 or 8 (per 16-bit A/B layout)

    for (int k0 = 0; k0 < K; k0 += TKs) {
        // ---- stage 128x32 A tile and 128x32 Bt tile into LDS (512 16B chunks each side) ----
#pragma unroll
        for (int it = 0; it < 2; ++it) {
            int c   = tid + it * 256;
            int row = c >> 2;
            int ko  = (c & 3) * 8;
#if USE_ASYNC_LDS_B128
            __builtin_amdgcn_global_load_async_to_lds_b128(
                ASG_V4I(&A [(size_t)(m0 + row) * K + k0 + ko]),
                ASL_V4I(&sA[row * LSTR + ko]), 0, 0);
            __builtin_amdgcn_global_load_async_to_lds_b128(
                ASG_V4I(&Bt[(size_t)(n0 + row) * K + k0 + ko]),
                ASL_V4I(&sB[row * LSTR + ko]), 0, 0);
#else
            *(v8bf*)&sA[row * LSTR + ko] = *(const v8bf*)&A [(size_t)(m0 + row) * K + k0 + ko];
            *(v8bf*)&sB[row * LSTR + ko] = *(const v8bf*)&Bt[(size_t)(n0 + row) * K + k0 + ko];
#endif
        }
#if USE_ASYNC_LDS_B128
        __builtin_amdgcn_s_wait_asynccnt(0);
#endif
        __syncthreads();

        // ---- load fragments (ISA 16-bit A layout: lanes0-15 K{0-7,16-23}, lanes16-31 K{8-15,24-31}) ----
        v16bf af[2], bfr[4];
#pragma unroll
        for (int mt = 0; mt < 2; ++mt) {
            const __bf16* p = &sA[(wm + mt * 16 + fr) * LSTR + kb];
            v8bf lo = *(const v8bf*)p;
            v8bf hi = *(const v8bf*)(p + 16);
            af[mt] = __builtin_shufflevector(lo, hi, 0,1,2,3,4,5,6,7,8,9,10,11,12,13,14,15);
        }
#pragma unroll
        for (int nt = 0; nt < 4; ++nt) {
            const __bf16* p = &sB[(wn + nt * 16 + fr) * LSTR + kb];
            v8bf lo = *(const v8bf*)p;
            v8bf hi = *(const v8bf*)(p + 16);
            bfr[nt] = __builtin_shufflevector(lo, hi, 0,1,2,3,4,5,6,7,8,9,10,11,12,13,14,15);
        }

        // ---- 8x v_wmma_f32_16x16x32_bf16 per wave per K-step ----
#pragma unroll
        for (int mt = 0; mt < 2; ++mt)
#pragma unroll
            for (int nt = 0; nt < 4; ++nt)
                acc[mt][nt] = __builtin_amdgcn_wmma_f32_16x16x32_bf16(
                    false, af[mt], false, bfr[nt], (short)0, acc[mt][nt], false, false);
        __syncthreads();
    }

    // ---- epilogue: C/D layout -> lane col = lane&15, row = e + 8*(lane>=16) ----
    const int cn    = lane & 15;
    const int rbase = (lane >> 4) * 8;
#pragma unroll
    for (int mt = 0; mt < 2; ++mt)
#pragma unroll
        for (int nt = 0; nt < 4; ++nt)
#pragma unroll
            for (int e = 0; e < 8; ++e) {
                int gr = m0 + wm + mt * 16 + rbase + e;
                int gc = n0 + wn + nt * 16 + cn;
                size_t idx = (size_t)gr * N + gc;
                float vv = acc[mt][nt][e];
                if (mode == 1) {
                    ((float*)Out)[idx] = vv + Res[idx];
                } else if (mode == 2) {
                    float rl = vv > 0.f ? vv : 0.f;
                    ((__bf16*)Out)[idx] = f2bf(rl * rl);
                } else {
                    ((float*)Out)[idx] = vv;
                }
            }
}

// ---------------- WKV6 recurrence: one block per (b,h), state [64x64] fp32 in regs ----------------
__global__ __launch_bounds__(256) void wkv6_kernel(
    const float* __restrict__ r, const float* __restrict__ k, const float* __restrict__ v,
    const float* __restrict__ time_decay, const float* __restrict__ time_first,
    float* __restrict__ wkv)
{
    __shared__ float ds[64], us[64], rsh[64], ksh[64], vsh[64], red[4][64];
    int bh = blockIdx.x;
    int b = bh / Hq, h = bh % Hq;
    int tid = threadIdx.x;
    int j  = tid & 63;      // v-dimension index owned by this thread
    int ig = tid >> 6;      // which 16-slice of the k-dimension

    if (tid < 64) {
        float w = time_decay[h * Sq + tid];
        ds[tid] = __expf(-__expf(w));     // decay in (0,1)
        us[tid] = time_first[h * Sq + tid];
    }
    __syncthreads();

    float st[16], dl[16], ul[16];
#pragma unroll
    for (int ii = 0; ii < 16; ++ii) {
        st[ii] = 0.f;
        dl[ii] = ds[ig * 16 + ii];
        ul[ii] = us[ig * 16 + ii];
    }

    size_t base = (size_t)b * Tq * Cq + (size_t)h * Sq;
    for (int tt = 0; tt < Tq; ++tt) {
        size_t off = base + (size_t)tt * Cq;
#if USE_ASYNC_LDS_B32
        if (tid < 64)
            __builtin_amdgcn_global_load_async_to_lds_b32(ASG_I32(&r[off + tid]),
                                                          ASL_I32(&rsh[tid]), 0, 0);
        else if (tid < 128)
            __builtin_amdgcn_global_load_async_to_lds_b32(ASG_I32(&k[off + tid - 64]),
                                                          ASL_I32(&ksh[tid - 64]), 0, 0);
        else if (tid < 192)
            __builtin_amdgcn_global_load_async_to_lds_b32(ASG_I32(&v[off + tid - 128]),
                                                          ASL_I32(&vsh[tid - 128]), 0, 0);
        __builtin_amdgcn_s_wait_asynccnt(0);
#else
        if (tid < 64)       rsh[tid]       = r[off + tid];
        else if (tid < 128) ksh[tid - 64]  = k[off + tid - 64];
        else if (tid < 192) vsh[tid - 128] = v[off + tid - 128];
#endif
        __syncthreads();

        float vj = vsh[j];
        float y  = 0.f;
#pragma unroll
        for (int ii = 0; ii < 16; ++ii) {
            int i = ig * 16 + ii;
            float kv = ksh[i] * vj;
            y  += rsh[i] * (st[ii] + ul[ii] * kv);
            st[ii] = st[ii] * dl[ii] + kv;
        }
        red[ig][j] = y;
        __syncthreads();
        if (tid < 64)
            wkv[off + j] = red[0][j] + red[1][j] + red[2][j] + red[3][j];
        // next iteration's LDS writes are separated by the barrier at loop top
    }
}

// ---------------- rwkv = sigmoid(r) * wkv -> bf16 ----------------
__global__ void sigmul_kernel(const float* __restrict__ r, const float* __restrict__ wkv,
                              __bf16* __restrict__ out) {
    size_t i = (size_t)blockIdx.x * 256 + threadIdx.x;
    float rr = r[i];
    float sg = 1.f / (1.f + __expf(-rr));
    out[i] = f2bf(sg * wkv[i]);
}

// ---------------- host-side orchestration ----------------
extern "C" void kernel_launch(void* const* d_in, const int* in_sizes, int n_in,
                              void* d_out, int out_size, void* d_ws, size_t ws_size,
                              hipStream_t stream) {
    const float* x    = (const float*)d_in[0];
    const float* tdec = (const float*)d_in[1];
    const float* tfir = (const float*)d_in[2];
    const float* w_r  = (const float*)d_in[3];
    const float* w_k  = (const float*)d_in[4];
    const float* w_v  = (const float*)d_in[5];
    const float* w_o  = (const float*)d_in[6];
    const float* w_fk = (const float*)d_in[7];
    const float* w_fv = (const float*)d_in[8];
    const float* g1   = (const float*)d_in[9];
    const float* b1   = (const float*)d_in[10];
    const float* g2   = (const float*)d_in[11];
    const float* b2   = (const float*)d_in[12];
    float* out = (float*)d_out;

    // carve workspace
    size_t off = 0;
    auto carve = [&](size_t bytes) {
        void* p = (char*)d_ws + off;
        off += (bytes + 255) & ~(size_t)255;
        return p;
    };
    __bf16* xln   = (__bf16*)carve((size_t)NTq * Cq * 2);
    __bf16* wrT   = (__bf16*)carve((size_t)Cq * Cq * 2);
    __bf16* wkT   = (__bf16*)carve((size_t)Cq * Cq * 2);
    __bf16* wvT   = (__bf16*)carve((size_t)Cq * Cq * 2);
    __bf16* woT   = (__bf16*)carve((size_t)Cq * Cq * 2);
    __bf16* wfkT  = (__bf16*)carve((size_t)Fq * Cq * 2);   // [F][C]
    __bf16* wfvT  = (__bf16*)carve((size_t)Cq * Fq * 2);   // [C][F]
    float*  rbuf  = (float*)carve((size_t)NTq * Cq * 4);
    float*  kbuf  = (float*)carve((size_t)NTq * Cq * 4);
    float*  vbuf  = (float*)carve((size_t)NTq * Cq * 4);
    float*  wkvb  = (float*)carve((size_t)NTq * Cq * 4);
    __bf16* rwkv  = (__bf16*)carve((size_t)NTq * Cq * 2);
    float*  x2    = (float*)carve((size_t)NTq * Cq * 4);
    __bf16* xln2  = (__bf16*)carve((size_t)NTq * Cq * 2);
    __bf16* hbuf  = (__bf16*)carve((size_t)NTq * Fq * 2);
    (void)ws_size; (void)in_sizes; (void)n_in; (void)out_size;

    dim3 tb(32, 8);
    // weight convert+transpose  (in [K][N] -> out [N][K] bf16)
    transpose_bf16_kernel<<<dim3(Cq / 32, Cq / 32), tb, 0, stream>>>(w_r,  wrT,  Cq, Cq);
    transpose_bf16_kernel<<<dim3(Cq / 32, Cq / 32), tb, 0, stream>>>(w_k,  wkT,  Cq, Cq);
    transpose_bf16_kernel<<<dim3(Cq / 32, Cq / 32), tb, 0, stream>>>(w_v,  wvT,  Cq, Cq);
    transpose_bf16_kernel<<<dim3(Cq / 32, Cq / 32), tb, 0, stream>>>(w_o,  woT,  Cq, Cq);
    transpose_bf16_kernel<<<dim3(Fq / 32, Cq / 32), tb, 0, stream>>>(w_fk, wfkT, Cq, Fq);
    transpose_bf16_kernel<<<dim3(Cq / 32, Fq / 32), tb, 0, stream>>>(w_fv, wfvT, Fq, Cq);

    // LN1
    ln_kernel<<<NTq, 256, 0, stream>>>(x, g1, b1, xln);

    // r/k/v projections (bf16 WMMA, fp32 out)
    dim3 gCC(Cq / TN, NTq / TM);
    gemm_bf16_kernel<<<gCC, 256, 0, stream>>>(xln, wrT, rbuf, nullptr, NTq, Cq, Cq, 0);
    gemm_bf16_kernel<<<gCC, 256, 0, stream>>>(xln, wkT, kbuf, nullptr, NTq, Cq, Cq, 0);
    gemm_bf16_kernel<<<gCC, 256, 0, stream>>>(xln, wvT, vbuf, nullptr, NTq, Cq, Cq, 0);

    // WKV recurrence over T
    wkv6_kernel<<<Bq * Hq, 256, 0, stream>>>(rbuf, kbuf, vbuf, tdec, tfir, wkvb);

    // gating + output projection with fused residual
    sigmul_kernel<<<(NTq * Cq) / 256, 256, 0, stream>>>(rbuf, wkvb, rwkv);
    gemm_bf16_kernel<<<gCC, 256, 0, stream>>>(rwkv, woT, x2, x, NTq, Cq, Cq, 1);

    // LN2 + FFN (relu^2 fused into GEMM1 epilogue; residual fused into GEMM2)
    ln_kernel<<<NTq, 256, 0, stream>>>(x2, g2, b2, xln2);
    dim3 gCF(Fq / TN, NTq / TM);
    gemm_bf16_kernel<<<gCF, 256, 0, stream>>>(xln2, wfkT, hbuf, nullptr, NTq, Fq, Cq, 2);
    gemm_bf16_kernel<<<gCC, 256, 0, stream>>>(hbuf, wfvT, out, x2, NTq, Cq, Fq, 1);
}
